// Conv2d_52561809768691
// MI455X (gfx1250) — matrix operand order repaired and checked by reading, not compiled, not run on hardware
//
#include <hip/hip_runtime.h>

typedef __attribute__((ext_vector_type(16))) _Float16 v16h;
typedef __attribute__((ext_vector_type(8)))  _Float16 v8h;
typedef __attribute__((ext_vector_type(2)))  _Float16 v2h;
typedef __attribute__((ext_vector_type(8)))  float    v8f;

#define CIN     128
#define COUT    256
#define WDIM    56
#define HW      3136        // 56*56
#define KTOT    1152        // CIN * 9
#define KT      32          // WMMA K depth per slice
#define NSLICE  36          // KTOT / KT ; slice s: tap r = s>>2, ci block cc = s&3
#define TILE_P  128         // pixels per workgroup
#define TILE_CO 64          // output channels per workgroup
#define ASTRIDE 40          // halves; 80B row stride (16B aligned, bank-staggered)
#define BSTRIDE 40
#define AHALVES (TILE_CO * ASTRIDE)
#define BHALVES (TILE_P * BSTRIDE)

// Build a v16h fragment from two contiguous 8-half runs in LDS.
__device__ __forceinline__ v16h load_frag16(const _Float16* p_lo, const _Float16* p_hi) {
    v8h lo = *(const v8h*)p_lo;
    v8h hi = *(const v8h*)p_hi;
    return __builtin_shufflevector(lo, hi, 0, 1, 2, 3, 4, 5, 6, 7,
                                           8, 9, 10, 11, 12, 13, 14, 15);
}

// Fetch one K-slice into registers: 16 im2col values (clamped-address gather,
// branchless) + 8 weight values. Tap (ky,kx) is uniform over the slice.
__device__ __forceinline__ void fetch_slice(int s,
                                            const float* __restrict__ xpix, // x + n*CIN*HW + bk_base*HW
                                            const float* __restrict__ wbase, // w + (co0+aco)*KTOT
                                            int ak_sub, int poy, int pox,
                                            float (&bv)[16], float (&av)[8], bool& inb) {
    const int r  = s >> 2;            // 0..8 (uniform)
    const int cc = s & 3;             // ci block (uniform)
    const int ky = (r * 11) >> 5;     // r/3 for r in 0..8
    const int kx = r - 3 * ky;
    const int iy = poy + ky - 1;
    const int ix = pox + kx - 1;
    inb = ((unsigned)iy < (unsigned)WDIM) & ((unsigned)ix < (unsigned)WDIM);
    const int iyc = iy < 0 ? 0 : (iy > WDIM - 1 ? WDIM - 1 : iy);
    const int ixc = ix < 0 ? 0 : (ix > WDIM - 1 ? WDIM - 1 : ix);
    const float* xp = xpix + (size_t)(cc * 32) * HW + iyc * WDIM + ixc;
    #pragma unroll
    for (int j = 0; j < 16; ++j)               // contiguous k_local run: stride HW
        bv[j] = xp[(size_t)j * HW];
    const float* wp = wbase + (size_t)(cc * 32 + ak_sub) * 9 + r;
    #pragma unroll
    for (int j = 0; j < 8; ++j)                // ci + j -> stride 9
        av[j] = wp[(size_t)j * 9];
}

__global__ __launch_bounds__(256)
void conv3x3_wmma_f16(const float* __restrict__ x,
                      const float* __restrict__ w,
                      const float* __restrict__ bias,
                      float* __restrict__ out) {
    __shared__ __align__(32) _Float16 ldsA[2 * AHALVES]; // [buf][co_local][k_local]
    __shared__ __align__(32) _Float16 ldsB[2 * BHALVES]; // [buf][p_local][k_local]

    const int tid  = threadIdx.x;
    const int lane = tid & 31;
    const int wave = tid >> 5;
    const int wm   = wave & 3;   // cout sub-tile: rows wm*16 .. wm*16+15
    const int wn   = wave >> 2;  // pixel sub-tile: cols wn*64 .. wn*64+63

    const int p0  = blockIdx.x * TILE_P;
    const int co0 = blockIdx.y * TILE_CO;

    // ---- B staging: thread owns one pixel column, 16 CONTIGUOUS k_locals
    const int bp_local = tid & (TILE_P - 1);
    const int bk_base  = (tid >> 7) << 4;            // 0 or 16
    const int p        = p0 + bp_local;
    const int pn       = p / HW;
    const int prem     = p - pn * HW;
    const int poy      = prem / WDIM;
    const int pox      = prem - poy * WDIM;
    const float* xpix  = x + (size_t)pn * (CIN * HW) + (size_t)bk_base * HW;

    // ---- A staging: thread owns one co row, 8 CONTIGUOUS k_locals
    const int aco      = tid >> 2;                   // 0..63
    const int ak_sub   = (tid & 3) << 3;             // 0,8,16,24
    const float* wbase = w + (size_t)(co0 + aco) * KTOT;

    // ---- fragment geometry (ISA §7.12.2 16-bit layouts)
    const int arow = (wm << 4) + (lane & 15);
    const int akb  = (lane < 16) ? 0 : 8;            // A: K {0..7,16..23} / {8..15,24..31}
    const int bkb  = (lane < 16) ? 0 : 8;            // B: K {0..7,16..23} / {8..15,24..31}
    const int abase0 = arow * ASTRIDE + akb;
    const int bbase0 = ((wn << 6) + (lane & 15)) * BSTRIDE + bkb;

    v8f acc[4];
    acc[0] = (v8f){}; acc[1] = (v8f){}; acc[2] = (v8f){}; acc[3] = (v8f){};

    float bv[16];
    float av[8];
    bool  inb;
    fetch_slice(0, xpix, wbase, ak_sub, poy, pox, bv, av, inb);

    for (int s = 0; s < NSLICE; ++s) {
        const int buf  = s & 1;
        const int aoff = buf * AHALVES;
        const int boff = buf * BHALVES;

        // ---- commit staged slice to LDS: packed stores
        // B: 8x ds_store_b32 (pair of halves), branchless zero-pad via select
        #pragma unroll
        for (int j = 0; j < 16; j += 2) {
            v2h pk;
            pk[0] = (_Float16)(inb ? bv[j]     : 0.0f);
            pk[1] = (_Float16)(inb ? bv[j + 1] : 0.0f);
            *(v2h*)&ldsB[boff + bp_local * BSTRIDE + bk_base + j] = pk;
        }
        // A: 1x ds_store_b128 (8 packed halves)
        {
            v8h ah;
            #pragma unroll
            for (int j = 0; j < 8; ++j)
                ah[j] = (_Float16)av[j];
            *(v8h*)&ldsA[aoff + aco * ASTRIDE + ak_sub] = ah;
        }

        // single barrier per slice: orders store(buf) -> compute(buf); with two
        // buffers this also (transitively) orders compute(buf^1)@s-1 before
        // store(buf^1)@s+1 across waves.
        __syncthreads();

        // ---- preload ALL fragments into distinct registers (lets the compiler
        //      issue the ds_loads as one batch, then the 4 WMMAs back-to-back
        //      without per-WMMA dscnt-0 drains)
        const v16h afrag = load_frag16(&ldsA[aoff + abase0],
                                       &ldsA[aoff + abase0 + 16]);
        v16h bfrag[4];
        #pragma unroll
        for (int t = 0; t < 4; ++t) {
            const int bb = boff + bbase0 + (t << 4) * BSTRIDE;
            bfrag[t] = load_frag16(&ldsB[bb], &ldsB[bb + 16]);
        }

        // ---- prefetch next slice (global loads overlap with WMMA chain below)
        const int sn = (s + 1 < NSLICE) ? s + 1 : NSLICE - 1;
        fetch_slice(sn, xpix, wbase, ak_sub, poy, pox, bv, av, inb);

        // ---- 4 back-to-back WMMAs
        #pragma unroll
        for (int t = 0; t < 4; ++t)
            acc[t] = __builtin_amdgcn_wmma_f32_16x16x32_f16(
                false, afrag, false, bfrag[t], (short)0, acc[t], false, false);
    }

    // ---- epilogue: C layout -> lane holds col lane&15, rows v + 8*(lane>=16)
    const int rowoff = (lane >> 4) * 8;
    float bvals[8];
    #pragma unroll
    for (int v = 0; v < 8; ++v)
        bvals[v] = bias[co0 + (wm << 4) + rowoff + v];

    #pragma unroll
    for (int t = 0; t < 4; ++t) {
        const int col = lane & 15;
        const int pp  = p0 + (wn << 6) + (t << 4) + col;
        const int n   = pp / HW;
        const int rem = pp - n * HW;          // oy*56 + ox
        const size_t obase = (size_t)n * COUT * HW + rem;
        float vals[8];
        *(v8f*)vals = acc[t];
        #pragma unroll
        for (int v = 0; v < 8; ++v) {
            const int co = co0 + (wm << 4) + rowoff + v;
            out[obase + (size_t)co * HW] = vals[v] + bvals[v];
        }
    }
}

extern "C" void kernel_launch(void* const* d_in, const int* in_sizes, int n_in,
                              void* d_out, int out_size, void* d_ws, size_t ws_size,
                              hipStream_t stream) {
    const float* x    = (const float*)d_in[0];
    const float* w    = (const float*)d_in[1];
    const float* bias = (const float*)d_in[2];
    float* out        = (float*)d_out;

    const int npix = in_sizes[0] / CIN;        // N * H * W = 100352
    dim3 grid(npix / TILE_P, COUT / TILE_CO);  // (784, 4)
    conv3x3_wmma_f16<<<grid, dim3(256), 0, stream>>>(x, w, bias, out);
}
